// QuantQwen2AudioEncoderLayer_3564822856089
// MI455X (gfx1250) — 16-bit and 8-bit matrix operand order repaired and checked by reading, compiled, not run on hardware
//
#include <hip/hip_runtime.h>
#include <hip/hip_bf16.h>
#include <cstdint>


typedef __attribute__((ext_vector_type(2)))  int       i2;
typedef __attribute__((ext_vector_type(4)))  int       i4;
typedef __attribute__((ext_vector_type(8)))  int       v8i;
typedef __attribute__((ext_vector_type(8)))  float     v8f;
typedef __attribute__((ext_vector_type(16))) _Float16  v16h;

#define TOK    16384
#define EDIM   1280
#define HEADS  20
#define HD     64
#define FDIM   5120
#define BSZ    16
#define SEQ    1024
#define QKVDIM 3840

typedef __attribute__((address_space(3))) char lds_char_t;

static __device__ __forceinline__ unsigned lds_off32(const void* p) {
    return (unsigned)(unsigned long long)(lds_char_t*)(char*)p;
}

static __device__ __forceinline__ void async_b128(unsigned ldsoff, const void* g) {
    asm volatile("global_load_async_to_lds_b128 %0, %1, off"
                 :: "v"(ldsoff), "v"(g) : "memory");
}

static __device__ __forceinline__ void wait_async0() {
    asm volatile("s_wait_asynccnt 0x0" ::: "memory");
}

__global__ __launch_bounds__(256) void quant_weight_kernel(
    const float* __restrict__ W, signed char* __restrict__ qw,
    float* __restrict__ sw, int K)
{
    const int row = blockIdx.x;
    const int tid = threadIdx.x;
    const float* w = W + (size_t)row * K;
    __shared__ float red[256];

    float mx = 0.f;
    for (int i = tid; i < K; i += 256) mx = fmaxf(mx, fabsf(w[i]));
    red[tid] = mx; __syncthreads();
    for (int off = 128; off > 0; off >>= 1) {
        if (tid < off) red[tid] = fmaxf(red[tid], red[tid + off]);
        __syncthreads();
    }
    const float s   = red[0] / 127.0f;
    const float inv = (s > 0.f) ? (1.0f / s) : 0.f;
    if (tid == 0) sw[row] = s;
    signed char* q = qw + (size_t)row * K;
    for (int i = tid; i < K; i += 256) q[i] = (signed char)(int)rintf(w[i] * inv);
}

__global__ __launch_bounds__(256) void rmsnorm_quant_kernel(
    const float* __restrict__ X, const float* __restrict__ lw,
    const float* __restrict__ lb, signed char* __restrict__ qx,
    float* __restrict__ sx)
{
    const int t = blockIdx.x, tid = threadIdx.x;
    const float* x = X + (size_t)t * EDIM;
    __shared__ float y[EDIM];
    __shared__ float red[256];

    float ss = 0.f;
    for (int i = tid; i < EDIM; i += 256) { float v = x[i]; ss += v * v; }
    red[tid] = ss; __syncthreads();
    for (int off = 128; off > 0; off >>= 1) {
        if (tid < off) red[tid] += red[tid + off];
        __syncthreads();
    }
    const float r = rsqrtf(red[0] / (float)EDIM + 1e-6f);
    __syncthreads();

    float mx = 0.f;
    for (int i = tid; i < EDIM; i += 256) {
        float v = x[i] * r * lw[i] + lb[i];
        y[i] = v; mx = fmaxf(mx, fabsf(v));
    }
    red[tid] = mx; __syncthreads();
    for (int off = 128; off > 0; off >>= 1) {
        if (tid < off) red[tid] = fmaxf(red[tid], red[tid + off]);
        __syncthreads();
    }
    const float s   = red[0] / 127.0f;
    const float inv = (s > 0.f) ? (1.0f / s) : 0.f;
    if (tid == 0) sx[t] = s;
    signed char* q = qx + (size_t)t * EDIM;
    for (int i = tid; i < EDIM; i += 256) {
        float v = fminf(fmaxf(rintf(y[i] * inv), -127.f), 127.f);
        q[i] = (signed char)(int)v;
    }
}

__global__ __launch_bounds__(256) void quant_token_kernel(
    const float* __restrict__ X, signed char* __restrict__ qx,
    float* __restrict__ sx, int N, int do_gelu)
{
    const int t = blockIdx.x, tid = threadIdx.x;
    const float* x = X + (size_t)t * N;
    __shared__ float y[FDIM];
    __shared__ float red[256];

    float mx = 0.f;
    for (int i = tid; i < N; i += 256) {
        float v = x[i];
        if (do_gelu) v = 0.5f * v * (1.0f + erff(v * 0.70710678118654752f));
        y[i] = v; mx = fmaxf(mx, fabsf(v));
    }
    red[tid] = mx; __syncthreads();
    for (int off = 128; off > 0; off >>= 1) {
        if (tid < off) red[tid] = fmaxf(red[tid], red[tid + off]);
        __syncthreads();
    }
    const float s   = red[0] / 127.0f;
    const float inv = (s > 0.f) ? (1.0f / s) : 0.f;
    if (tid == 0) sx[t] = s;
    signed char* q = qx + (size_t)t * N;
    for (int i = tid; i < N; i += 256) {
        float v = fminf(fmaxf(rintf(y[i] * inv), -127.f), 127.f);
        q[i] = (signed char)(int)v;
    }
}

__global__ __launch_bounds__(256) void gemm_i8_kernel(
    const signed char* __restrict__ qA, const signed char* __restrict__ qB,
    const float* __restrict__ sA, const float* __restrict__ sB,
    const float* __restrict__ bias, const float* __restrict__ res,
    float* __restrict__ C, int M, int N, int K)
{
    const int lane = threadIdx.x & 31;
    const int wave = threadIdx.x >> 5;
    const int half = lane >> 4;
    const int l15  = lane & 15;

    const int m0 = blockIdx.y * 64 + (wave >> 1) * 16;
    const int n0 = blockIdx.x * 32 + (wave & 1) * 16;

    const signed char* aRow = qA + (size_t)(m0 + l15) * K + half * 16;
    const signed char* bCol = qB + (size_t)(n0 + l15) * K + half * 16;

    v8i acc = {};
    for (int k0 = 0; k0 < K; k0 += 64) {
        __builtin_prefetch(aRow + k0 + 64, 0, 3);
        __builtin_prefetch(bCol + k0 + 64, 0, 3);

        i2 a0 = *(const i2*)(aRow + k0);
        i2 a1 = *(const i2*)(aRow + k0 + 8);
        i2 a2 = *(const i2*)(aRow + k0 + 32);
        i2 a3 = *(const i2*)(aRow + k0 + 40);
        v8i a; a[0]=a0[0]; a[1]=a0[1]; a[2]=a1[0]; a[3]=a1[1];
               a[4]=a2[0]; a[5]=a2[1]; a[6]=a3[0]; a[7]=a3[1];

        i4 b0 = *(const i4*)(bCol + k0);
        i4 b1 = *(const i4*)(bCol + k0 + 32);
        v8i b; b[0]=b0[0]; b[1]=b0[1]; b[2]=b0[2]; b[3]=b0[3];
               b[4]=b1[0]; b[5]=b1[1]; b[6]=b1[2]; b[7]=b1[3];

        acc = __builtin_amdgcn_wmma_i32_16x16x64_iu8(
             true, a,  true, b, acc,
             false,  false);
    }

    const int  col = n0 + l15;
    const float swv = sB[col];
    const float bv  = bias[col];
#pragma unroll
    for (int r = 0; r < 8; ++r) {
        const int row = m0 + r + half * 8;
        float v = (float)acc[r] * (sA[row] * swv) + bv;
        const size_t idx = (size_t)row * N + col;
        if (res) v += res[idx];
        C[idx] = v;
    }
}

__global__ __launch_bounds__(256) void attention_kernel(
    const float* __restrict__ qkv, float* __restrict__ attn)
{
    __shared__ float    kslab[32 * HD];
    __shared__ float    vslab[32 * HD];
    __shared__ _Float16 pbuf[8][16 * 32];

    const int tid  = threadIdx.x;
    const int lane = tid & 31;
    const int wave = tid >> 5;
    const int half = lane >> 4;
    const int l15  = lane & 15;

    const int b  = blockIdx.x / HEADS;
    const int h  = blockIdx.x % HEADS;
    const int q0 = blockIdx.y * 128 + wave * 16;

    const size_t rowStride = (size_t)QKVDIM;
    const float* Qb = qkv + (size_t)b * SEQ * rowStride + (size_t)h * HD;
    const float* Kb = Qb + EDIM;
    const float* Vb = Qb + 2 * EDIM;

    const int srow = tid >> 3;
    const int scol = (tid & 7) * 8;
    const float* kg0 = Kb + (size_t)srow * rowStride + scol;
    const float* vg0 = Vb + (size_t)srow * rowStride + scol;
    const unsigned klo = lds_off32(&kslab[srow * HD + scol]);
    const unsigned vlo = lds_off32(&vslab[srow * HD + scol]);

    _Float16* P = &pbuf[wave][0];

    v16h qa0, qa1;
    {
        const float* qrow = Qb + (size_t)(q0 + l15) * rowStride + half * 16;
#pragma unroll
        for (int j = 0; j < 8; ++j) {
            qa0[j]     = (_Float16)qrow[j];
            qa0[8 + j] = (_Float16)qrow[8 + j];
            qa1[j]     = (_Float16)qrow[32 + j];
            qa1[8 + j] = (_Float16)qrow[40 + j];
        }
    }

    v8f o0 = {}, o1 = {}, o2 = {}, o3 = {};
    float mrow[8], lrow[8];
#pragma unroll
    for (int r = 0; r < 8; ++r) { mrow[r] = -1e30f; lrow[r] = 0.f; }

    for (int j0 = 0; j0 < SEQ; j0 += 32) {
        __syncthreads();
        const size_t jof = (size_t)j0 * rowStride;
        async_b128(klo,      kg0 + jof);
        async_b128(klo + 16, kg0 + jof + 4);
        async_b128(vlo,      vg0 + jof);
        async_b128(vlo + 16, vg0 + jof + 4);
        wait_async0();
        __syncthreads();

        v8f st[2];
#pragma unroll
        for (int ct = 0; ct < 2; ++ct) {
            const float* krow = &kslab[(16 * ct + l15) * HD + half * 16];
            v16h kf0, kf1;
#pragma unroll
            for (int j = 0; j < 16; ++j) {
                kf0[j] = (_Float16)krow[j];
                kf1[j] = (_Float16)krow[32 + j];
            }
            v8f s = {};
            s = __builtin_amdgcn_wmma_f32_16x16x32_f16(false, qa0, false, kf0,
                                                       (short)0, s, false, false);
            s = __builtin_amdgcn_wmma_f32_16x16x32_f16(false, qa1, false, kf1,
                                                       (short)0, s, false, false);
            st[ct] = s * 0.125f;
        }

#pragma unroll
        for (int r = 0; r < 8; ++r) {
            float rv = fmaxf(st[0][r], st[1][r]);
#pragma unroll
            for (int off = 8; off > 0; off >>= 1)
                rv = fmaxf(rv, __shfl_xor(rv, off, 16));
            const float nm   = fmaxf(mrow[r], rv);
            const float corr = __expf(mrow[r] - nm);
            const float p0   = __expf(st[0][r] - nm);
            const float p1   = __expf(st[1][r] - nm);
            float rs = p0 + p1;
#pragma unroll
            for (int off = 8; off > 0; off >>= 1)
                rs += __shfl_xor(rs, off, 16);
            lrow[r] = lrow[r] * corr + rs;
            mrow[r] = nm;
            o0[r] *= corr; o1[r] *= corr; o2[r] *= corr; o3[r] *= corr;
            const int row = r + 8 * half;
            P[row * 32 + l15]      = (_Float16)p0;
            P[row * 32 + 16 + l15] = (_Float16)p1;
        }

        v16h pa;
        {
            const _Float16* pr = P + l15 * 32 + half * 8;
#pragma unroll
            for (int j = 0; j < 8; ++j) {
                pa[j]     = pr[j];
                pa[8 + j] = pr[16 + j];
            }
        }

#pragma unroll
        for (int nc = 0; nc < 4; ++nc) {
            v16h vf;
#pragma unroll
            for (int j = 0; j < 16; ++j)
                vf[j] = (_Float16)vslab[(half * 8 + j + (j & 8)) * HD + nc * 16 + l15];
            v8f* op = (nc == 0) ? &o0 : (nc == 1) ? &o1 : (nc == 2) ? &o2 : &o3;
            *op = __builtin_amdgcn_wmma_f32_16x16x32_f16(false, pa, false, vf,
                                                         (short)0, *op, false, false);
        }
    }

#pragma unroll
    for (int r = 0; r < 8; ++r) {
        const float inv = 1.0f / lrow[r];
        const int   row = q0 + r + 8 * half;
        float* orow = attn + ((size_t)b * SEQ + row) * EDIM + (size_t)h * HD;
        orow[l15]      = o0[r] * inv;
        orow[16 + l15] = o1[r] * inv;
        orow[32 + l15] = o2[r] * inv;
        orow[48 + l15] = o3[r] * inv;
    }
}

static inline size_t ws_alloc(size_t& off, size_t bytes) {
    size_t o = off;
    off = (off + bytes + 255) & ~(size_t)255;
    return o;
}

extern "C" void kernel_launch(void* const* d_in, const int* in_sizes, int n_in,
                              void* d_out, int out_size, void* d_ws, size_t ws_size,
                              hipStream_t stream) {
    (void)in_sizes; (void)n_in; (void)out_size; (void)ws_size;

    const float* hidden = (const float*)d_in[0];
    const float* ln1_w  = (const float*)d_in[1];
    const float* ln1_b  = (const float*)d_in[2];
    const float* ln2_w  = (const float*)d_in[3];
    const float* ln2_b  = (const float*)d_in[4];
    const float* W_qkv  = (const float*)d_in[5];
    const float* b_qkv  = (const float*)d_in[6];
    const float* W_out  = (const float*)d_in[7];
    const float* b_out  = (const float*)d_in[8];
    const float* W_fc1  = (const float*)d_in[9];
    const float* b_fc1  = (const float*)d_in[10];
    const float* W_fc2  = (const float*)d_in[11];
    const float* b_fc2  = (const float*)d_in[12];
    float* out = (float*)d_out;

    char* ws = (char*)d_ws;
    size_t off = 0;
    signed char* qw_qkv = (signed char*)(ws + ws_alloc(off, (size_t)QKVDIM * EDIM));
    signed char* qw_out = (signed char*)(ws + ws_alloc(off, (size_t)EDIM * EDIM));
    signed char* qw_fc1 = (signed char*)(ws + ws_alloc(off, (size_t)FDIM * EDIM));
    signed char* qw_fc2 = (signed char*)(ws + ws_alloc(off, (size_t)EDIM * FDIM));
    float* sw_qkv = (float*)(ws + ws_alloc(off, (size_t)QKVDIM * 4));
    float* sw_out = (float*)(ws + ws_alloc(off, (size_t)EDIM * 4));
    float* sw_fc1 = (float*)(ws + ws_alloc(off, (size_t)FDIM * 4));
    float* sw_fc2 = (float*)(ws + ws_alloc(off, (size_t)EDIM * 4));
    signed char* qact = (signed char*)(ws + ws_alloc(off, (size_t)TOK * FDIM));
    float* sx   = (float*)(ws + ws_alloc(off, (size_t)TOK * 4));
    float* qkv  = (float*)(ws + ws_alloc(off, (size_t)TOK * QKVDIM * 4));
    float* attn = (float*)(ws + ws_alloc(off, (size_t)TOK * EDIM * 4));
    float* hid1 = (float*)(ws + ws_alloc(off, (size_t)TOK * EDIM * 4));
    float* f1   = (float*)(ws + ws_alloc(off, (size_t)TOK * FDIM * 4));

    quant_weight_kernel<<<QKVDIM, 256, 0, stream>>>(W_qkv, qw_qkv, sw_qkv, EDIM);
    quant_weight_kernel<<<EDIM,   256, 0, stream>>>(W_out, qw_out, sw_out, EDIM);
    quant_weight_kernel<<<FDIM,   256, 0, stream>>>(W_fc1, qw_fc1, sw_fc1, EDIM);
    quant_weight_kernel<<<EDIM,   256, 0, stream>>>(W_fc2, qw_fc2, sw_fc2, FDIM);

    rmsnorm_quant_kernel<<<TOK, 256, 0, stream>>>(hidden, ln1_w, ln1_b, qact, sx);
    gemm_i8_kernel<<<dim3(QKVDIM / 32, TOK / 64), 256, 0, stream>>>(
        qact, qw_qkv, sx, sw_qkv, b_qkv, nullptr, qkv, TOK, QKVDIM, EDIM);
    attention_kernel<<<dim3(BSZ * HEADS, SEQ / 128), 256, 0, stream>>>(qkv, attn);
    quant_token_kernel<<<TOK, 256, 0, stream>>>(attn, qact, sx, EDIM, 0);
    gemm_i8_kernel<<<dim3(EDIM / 32, TOK / 64), 256, 0, stream>>>(
        qact, qw_out, sx, sw_out, b_out, hidden, hid1, TOK, EDIM, EDIM);

    rmsnorm_quant_kernel<<<TOK, 256, 0, stream>>>(hid1, ln2_w, ln2_b, qact, sx);
    gemm_i8_kernel<<<dim3(FDIM / 32, TOK / 64), 256, 0, stream>>>(
        qact, qw_fc1, sx, sw_fc1, b_fc1, nullptr, f1, TOK, FDIM, EDIM);
    quant_token_kernel<<<TOK, 256, 0, stream>>>(f1, qact, sx, FDIM, 1);
    gemm_i8_kernel<<<dim3(EDIM / 32, TOK / 64), 256, 0, stream>>>(
        qact, qw_fc2, sx, sw_fc2, b_fc2, hid1, out, TOK, EDIM, FDIM);
}
